// GCN2_64699387347696
// MI455X (gfx1250) — compile-verified, matrix operand order checked by reading, not run on hardware
//
#include <hip/hip_runtime.h>
#include <hip/hip_bf16.h>

typedef __attribute__((ext_vector_type(16))) _Float16 v16h;
typedef __attribute__((ext_vector_type(8)))  float    v8f;
typedef __attribute__((ext_vector_type(4)))  _Float16 h4;

#define DD 128
#define CC 64
#define DELTA 0.01f

// ---------------- setup kernels ----------------

__global__ void zero_kernel(float* deg, int* cnt, int* fpos, int n) {
    int i = blockIdx.x * blockDim.x + threadIdx.x;
    if (i < n) { deg[i] = 0.0f; cnt[i] = 0; fpos[i] = 0; }
}

__global__ void deg_count_kernel(const int* __restrict__ ei, const float* __restrict__ ew,
                                 float* deg, int* cnt, int e_total) {
    int e = blockIdx.x * blockDim.x + threadIdx.x;
    if (e >= e_total) return;
    int r = ei[e];                       // edge_index[0][e]
    atomicAdd(&deg[r], ew[e]);
    atomicAdd(&cnt[r], 1);
}

// single-block exclusive scan: rptr[0..n], rptr[n] = total
__global__ void scan_kernel(const int* __restrict__ cnt, int* __restrict__ rptr, int n) {
    __shared__ int sh[1024];
    __shared__ int carry;
    if (threadIdx.x == 0) carry = 0;
    __syncthreads();
    for (int base = 0; base < n; base += 1024) {
        int i = base + (int)threadIdx.x;
        int v = (i < n) ? cnt[i] : 0;
        sh[threadIdx.x] = v;
        __syncthreads();
        for (int off = 1; off < 1024; off <<= 1) {
            int t = (threadIdx.x >= (unsigned)off) ? sh[threadIdx.x - off] : 0;
            __syncthreads();
            sh[threadIdx.x] += t;
            __syncthreads();
        }
        int incl = sh[threadIdx.x];
        if (i < n) rptr[i] = carry + (incl - v);
        __syncthreads();
        if (threadIdx.x == 1023) carry += sh[1023];
        __syncthreads();
    }
    if (threadIdx.x == 0) rptr[n] = carry;
}

__global__ void fill_kernel(const int* __restrict__ ei, const float* __restrict__ ew,
                            const float* __restrict__ deg, const int* __restrict__ rptr,
                            int* fpos, int* __restrict__ ccol, float* __restrict__ cw,
                            int e_total) {
    int e = blockIdx.x * blockDim.x + threadIdx.x;
    if (e >= e_total) return;
    int r = ei[e];
    int c = ei[e_total + e];             // edge_index[1][e]
    int p = atomicAdd(&fpos[r], 1);
    int idx = rptr[r] + p;
    ccol[idx] = c;
    float d = deg[r];
    cw[idx] = (d > 0.0f) ? (ew[e] / d) : 0.0f;   // w_norm in CSR order
}

// ---------------- per-node trig cache: cx/sx in f16 (node-level, not edge-level) ----------------

__global__ void __launch_bounds__(256)
trig_kernel(const float* __restrict__ xc, _Float16* __restrict__ cxh,
            _Float16* __restrict__ sxh, size_t total4) {
    size_t t = (size_t)blockIdx.x * blockDim.x + threadIdx.x;
    if (t >= total4) return;
    size_t base = t * 4;
    float4 x = *(const float4*)(xc + base);
    h4 c, s;
    c[0] = (_Float16)__cosf(x.x); s[0] = (_Float16)__sinf(x.x);
    c[1] = (_Float16)__cosf(x.y); s[1] = (_Float16)__sinf(x.y);
    c[2] = (_Float16)__cosf(x.z); s[2] = (_Float16)__sinf(x.z);
    c[3] = (_Float16)__cosf(x.w); s[3] = (_Float16)__sinf(x.w);
    *(h4*)(cxh + base) = c;
    *(h4*)(sxh + base) = s;
}

// ---------------- fused aggregation + update (one wave per node) ----------------

__global__ void __launch_bounds__(256)
agg_kernel(const float* __restrict__ xc, float* __restrict__ xn,
           const float* __restrict__ orig,
           const _Float16* __restrict__ cxh, const _Float16* __restrict__ sxh,
           const int* __restrict__ rptr, const int* __restrict__ ccol,
           const float* __restrict__ cw, int n) {
    int wave = blockIdx.x * (blockDim.x >> 5) + (threadIdx.x >> 5);
    int lane = threadIdx.x & 31;
    if (wave >= n) return;

    size_t base = (size_t)wave * DD + 4 * lane;       // lane owns 4 contiguous cols
    float4 xi = *(const float4*)(xc + base);
    float4 oi = *(const float4*)(orig + base);

    float4 aC = make_float4(0.f, 0.f, 0.f, 0.f);
    float4 aS = make_float4(0.f, 0.f, 0.f, 0.f);
    float4 aX = make_float4(0.f, 0.f, 0.f, 0.f);

    int s = rptr[wave], e = rptr[wave + 1];
    for (int k = s; k < e; ++k) {
        if (k + 8 < e) {
            int jp = ccol[k + 8];
            __builtin_prefetch(xc + (size_t)jp * DD + 4 * lane, 0, 0);
        }
        int   j = ccol[k];
        float w = cw[k];
        size_t jb = (size_t)j * DD + 4 * lane;
        float4 xj = *(const float4*)(xc + jb);
        h4 cj = *(const h4*)(cxh + jb);
        h4 sj = *(const h4*)(sxh + jb);
        aC.x += w * (float)cj[0]; aS.x += w * (float)sj[0]; aX.x += w * xj.x;
        aC.y += w * (float)cj[1]; aS.y += w * (float)sj[1]; aX.y += w * xj.y;
        aC.z += w * (float)cj[2]; aS.z += w * (float)sj[2]; aX.z += w * xj.z;
        aC.w += w * (float)cj[3]; aS.w += w * (float)sj[3]; aX.w += w * xj.w;
    }

    // cx - cos_R == agg_c,  sin(-phi) == sin(agg_x - x)
    // R = sqrt(agg_c^2 + agg_s^2);  x += DELTA*(orig + R*sin(agg_x - x))
    float4 xo;
    xo.x = xi.x + DELTA * (oi.x + __fsqrt_rn(aC.x * aC.x + aS.x * aS.x) * __sinf(aX.x - xi.x));
    xo.y = xi.y + DELTA * (oi.y + __fsqrt_rn(aC.y * aC.y + aS.y * aS.y) * __sinf(aX.y - xi.y));
    xo.z = xi.z + DELTA * (oi.z + __fsqrt_rn(aC.z * aC.z + aS.z * aS.z) * __sinf(aX.z - xi.z));
    xo.w = xi.w + DELTA * (oi.w + __fsqrt_rn(aC.w * aC.w + aS.w * aS.w) * __sinf(aX.w - xi.w));
    *(float4*)(xn + base) = xo;
}

// ---------------- final projection: sigmoid(relu(x) @ W^T + b) via WMMA ----------------
// One wave computes a 16(M) x 64(N) tile: 1 A operand, 4 B operands, 4 v8f accumulators.

__global__ void __launch_bounds__(128)
gemm_kernel(const float* __restrict__ x, const float* __restrict__ w,
            const float* __restrict__ bias, float* __restrict__ out, int n) {
    int wave = blockIdx.x * (blockDim.x >> 5) + (threadIdx.x >> 5);
    int m0 = wave * 16;
    if (m0 >= n) return;                 // wave-uniform: EXEC stays all-ones for WMMA
    int lane = threadIdx.x & 31;
    int l16  = lane & 15;
    int half = lane >> 4;
    int kb   = half * 8;

    v8f acc[4] = {v8f{}, v8f{}, v8f{}, v8f{}};
    const float* xr = x + (size_t)(m0 + l16) * DD;

    for (int k0 = 0; k0 < DD; k0 += 32) {
        // A operand: 16x32 f16, ISA layout (elems 0-7 -> K=kb..kb+7, 8-15 -> K=kb+16..kb+23)
        v16h a;
#pragma unroll
        for (int i = 0; i < 8; ++i) {
            float v = xr[k0 + kb + i];
            a[i] = (_Float16)(v > 0.0f ? v : 0.0f);      // fused relu
        }
#pragma unroll
        for (int i = 0; i < 8; ++i) {
            float v = xr[k0 + 16 + kb + i];
            a[8 + i] = (_Float16)(v > 0.0f ? v : 0.0f);
        }
#pragma unroll
        for (int ct = 0; ct < 4; ++ct) {
            const float* wr = w + (size_t)(ct * 16 + l16) * DD;  // lin_w CxD row-major; B[k,c]=W[c,k]
            v16h bb;
#pragma unroll
            for (int i = 0; i < 8; ++i) bb[i]     = (_Float16)wr[k0 + kb + i];
#pragma unroll
            for (int i = 0; i < 8; ++i) bb[8 + i] = (_Float16)wr[k0 + 16 + kb + i];
            acc[ct] = __builtin_amdgcn_wmma_f32_16x16x32_f16(
                false, a, false, bb, (short)0, acc[ct], false, false);
        }
    }

    // C/D layout: VGPR v -> M = v + 8*half, N = l16 (per 16-wide column tile)
#pragma unroll
    for (int ct = 0; ct < 4; ++ct) {
        int ncol = ct * 16 + l16;
        float bv = bias[ncol];
#pragma unroll
        for (int v = 0; v < 8; ++v) {
            int m = m0 + v + 8 * half;
            float val = acc[ct][v] + bv;
            out[(size_t)m * CC + ncol] = 1.0f / (1.0f + __expf(-val));
        }
    }
}

// ---------------- host ----------------

static inline size_t align_up(size_t v, size_t a) { return (v + a - 1) & ~(a - 1); }

extern "C" void kernel_launch(void* const* d_in, const int* in_sizes, int n_in,
                              void* d_out, int out_size, void* d_ws, size_t ws_size,
                              hipStream_t stream) {
    const int*   edge_index  = (const int*)d_in[0];    // (2, E) row-major
    const float* edge_weight = (const float*)d_in[1];  // (E,)
    const float* embed       = (const float*)d_in[2];  // (N, 128)
    const float* lin_w       = (const float*)d_in[3];  // (64, 128)
    const float* lin_b       = (const float*)d_in[4];  // (64,)
    const int E = in_sizes[1];
    const int N = in_sizes[2] / DD;

    // workspace carve-up
    char* ws = (char*)d_ws;
    size_t off = 0;
    float*     x0   = (float*)(ws + off);     off = align_up(off + (size_t)N * DD * 4, 256);
    float*     x1   = (float*)(ws + off);     off = align_up(off + (size_t)N * DD * 4, 256);
    _Float16*  cxh  = (_Float16*)(ws + off);  off = align_up(off + (size_t)N * DD * 2, 256);
    _Float16*  sxh  = (_Float16*)(ws + off);  off = align_up(off + (size_t)N * DD * 2, 256);
    float*     deg  = (float*)(ws + off);     off = align_up(off + (size_t)N * 4, 256);
    int*       cnt  = (int*)  (ws + off);     off = align_up(off + (size_t)N * 4, 256);
    int*       fpos = (int*)  (ws + off);     off = align_up(off + (size_t)N * 4, 256);
    int*       rptr = (int*)  (ws + off);     off = align_up(off + ((size_t)N + 1) * 4, 256);
    int*       ccol = (int*)  (ws + off);     off = align_up(off + (size_t)E * 4, 256);
    float*     cw   = (float*)(ws + off);     off = align_up(off + (size_t)E * 4, 256);

    float* out_probs = (float*)d_out;                     // N*64
    float* out_x     = (float*)d_out + (size_t)N * CC;    // N*128

    // CSR build (per call, deterministic work)
    zero_kernel<<<(N + 255) / 256, 256, 0, stream>>>(deg, cnt, fpos, N);
    deg_count_kernel<<<(E + 255) / 256, 256, 0, stream>>>(edge_index, edge_weight, deg, cnt, E);
    scan_kernel<<<1, 1024, 0, stream>>>(cnt, rptr, N);
    fill_kernel<<<(E + 255) / 256, 256, 0, stream>>>(edge_index, edge_weight, deg, rptr,
                                                     fpos, ccol, cw, E);

    // x <- embed_weight
    hipMemcpyAsync(x0, embed, (size_t)N * DD * sizeof(float), hipMemcpyDeviceToDevice, stream);

    float* cur = x0;
    float* nxt = x1;
    const size_t total4 = (size_t)N * DD / 4;
    const int trig_blocks = (int)((total4 + 255) / 256);
    const int waves_per_block = 256 / 32;
    const int prop_blocks = (N + waves_per_block - 1) / waves_per_block;
    for (int step = 0; step < 20; ++step) {
        trig_kernel<<<trig_blocks, 256, 0, stream>>>(cur, cxh, sxh, total4);
        agg_kernel<<<prop_blocks, 256, 0, stream>>>(cur, nxt, embed, cxh, sxh,
                                                    rptr, ccol, cw, N);
        float* t = cur; cur = nxt; nxt = t;
    }

    // output x
    hipMemcpyAsync(out_x, cur, (size_t)N * DD * sizeof(float), hipMemcpyDeviceToDevice, stream);

    // output probs via WMMA GEMM
    const int mtiles = (N + 15) / 16;                 // 3125 waves
    const int gemm_blocks = (mtiles + 3) / 4;         // 4 waves / block
    gemm_kernel<<<gemm_blocks, 128, 0, stream>>>(cur, lin_w, lin_b, out_probs, N);
}
